// WebObjExtractionNet_13864154431526
// MI455X (gfx1250) — matrix operand order repaired and checked by reading, not compiled, not run on hardware
//
#include <hip/hip_runtime.h>
#include <hip/hip_bf16.h>

typedef __bf16 bf16_t;
typedef __attribute__((ext_vector_type(16))) __bf16 v16bf;
typedef __attribute__((ext_vector_type(8)))  __bf16 v8bf;
typedef __attribute__((ext_vector_type(8)))  float  v8f;

#define NB    1024
#define K2C   10
#define CF    384
#define FEAT  13
#define NVIS  3456      // 384*9, multiple of 32
#define HIDN  300
#define HIDNP64 320     // ceil64(300): B-row padding for 4-tile GEMM
#define HKP   320       // K-pad for second encoder layer (mult of 32)
#define NTOT  3760
#define NTOTP 3776      // K-pad (mult of 32) == ceil64(3760)

// ---------------------------------------------------------------- conv (f32)
__global__ __launch_bounds__(256)
void conv2d_k(const float* __restrict__ x, const float* __restrict__ w,
              const float* __restrict__ bias, float* __restrict__ y,
              int Nn, int IC, int IH, int IW, int OC, int OH, int OW,
              int KH, int KW, int stride, int pad, int relu) {
  int idx = blockIdx.x * blockDim.x + threadIdx.x;
  int total = Nn * OC * OH * OW;
  if (idx >= total) return;
  int ow = idx % OW; int t = idx / OW;
  int oh = t % OH;   t /= OH;
  int oc = t % OC;   int n = t / OC;
  float acc = bias[oc];
  int ih0 = oh * stride - pad, iw0 = ow * stride - pad;
  for (int ic = 0; ic < IC; ++ic) {
    const float* xp = x + ((size_t)(n * IC + ic) * IH) * IW;
    const float* wp = w + ((size_t)(oc * IC + ic) * KH) * KW;
    for (int kh = 0; kh < KH; ++kh) {
      int ih = ih0 + kh; if ((unsigned)ih >= (unsigned)IH) continue;
      for (int kw = 0; kw < KW; ++kw) {
        int iw = iw0 + kw; if ((unsigned)iw >= (unsigned)IW) continue;
        acc = fmaf(xp[ih * IW + iw], wp[kh * KW + kw], acc);
      }
    }
  }
  y[idx] = relu ? fmaxf(acc, 0.f) : acc;
}

// ---------------------------------------------------------------- maxpool 3x3 s2
__global__ __launch_bounds__(256)
void maxpool3s2_k(const float* __restrict__ x, float* __restrict__ y,
                  int NC, int H, int W, int HO, int WO) {
  int idx = blockIdx.x * blockDim.x + threadIdx.x;
  int total = NC * HO * WO;
  if (idx >= total) return;
  int wo = idx % WO; int t = idx / WO;
  int ho = t % HO;   int nc = t / HO;
  const float* xp = x + (size_t)nc * H * W;
  float m = -__builtin_inff();
  for (int kh = 0; kh < 3; ++kh)
    for (int kw = 0; kw < 3; ++kw)
      m = fmaxf(m, xp[(ho * 2 + kh) * W + (wo * 2 + kw)]);
  y[idx] = m;
}

// ---------------------------------------------------------------- ROI max-pool
__global__ __launch_bounds__(256)
void roi_pool_k(const float* __restrict__ feat, const float* __restrict__ bb,
                float* __restrict__ own) {
  int idx = blockIdx.x * blockDim.x + threadIdx.x;
  if (idx >= NB * CF) return;
  int c = idx % CF, n = idx / CF;
  const float S = (float)FEAT / 224.0f;
  int img = (int)bb[n * 5 + 0];
  int x1 = (int)rintf(bb[n * 5 + 1] * S);
  int y1 = (int)rintf(bb[n * 5 + 2] * S);
  int x2 = (int)rintf(bb[n * 5 + 3] * S);
  int y2 = (int)rintf(bb[n * 5 + 4] * S);
  int rw = max(x2 - x1 + 1, 1), rh = max(y2 - y1 + 1, 1);
  const float* f = feat + ((size_t)img * CF + c) * FEAT * FEAT;
  for (int ph = 0; ph < 3; ++ph) {
    int hs = min(max(y1 + (ph * rh) / 3, 0), FEAT);
    int he = min(max(y1 + ((ph + 1) * rh + 2) / 3, 0), FEAT);
    for (int pw = 0; pw < 3; ++pw) {
      int ws_ = min(max(x1 + (pw * rw) / 3, 0), FEAT);
      int we  = min(max(x1 + ((pw + 1) * rw + 2) / 3, 0), FEAT);
      float v = 0.f;
      if (he > hs && we > ws_) {
        v = -__builtin_inff();
        for (int h = hs; h < he; ++h)
          for (int w = ws_; w < we; ++w)
            v = fmaxf(v, f[h * FEAT + w]);
      }
      own[(size_t)n * NVIS + c * 9 + ph * 3 + pw] = v;
    }
  }
}

// ---------------------------------------------------------------- helpers
__global__ __launch_bounds__(256)
void cvt_f32_bf16_k(const float* __restrict__ src, bf16_t* __restrict__ dst, int n) {
  int i = blockIdx.x * blockDim.x + threadIdx.x;
  if (i < n) dst[i] = (bf16_t)src[i];
}

__global__ __launch_bounds__(256)
void fill_bf16_zero_k(bf16_t* __restrict__ p, int n) {
  int i = blockIdx.x * blockDim.x + threadIdx.x;
  if (i < n) p[i] = (bf16_t)0.f;
}

// W (KxN row-major, f32) -> WT (Np x Kp row-major, bf16, zero-padded)
__global__ __launch_bounds__(256)
void wT_bf16_k(const float* __restrict__ w, bf16_t* __restrict__ wt,
               int K, int N, int Kp, int Np) {
  int idx = blockIdx.x * blockDim.x + threadIdx.x;
  int total = Np * Kp;
  if (idx >= total) return;
  int k = idx % Kp, n = idx / Kp;
  wt[idx] = (n < N && k < K) ? (bf16_t)w[(size_t)k * N + n] : (bf16_t)0.f;
}

// ---------------------------------------------------------------- WMMA GEMM
// C[M,N] = A[M,Kp](bf16) x B^T, B given as WT[ceil64(N) x Kp](bf16).
// Each wave: one 16-row M tile x four 16-col N tiles (4 accumulators, shared A frag).
// Row gather: idx<0 -> reads pre-zeroed dummy row (no predication in K loop).
__device__ __forceinline__ v16bf load_a_frag(const bf16_t* p) {
  return *(const v16bf*)p;            // K = base .. base+15
}

#define WMMA_BF16(a, b, c) __builtin_amdgcn_wmma_f32_16x16x32_bf16( \
    false, (a), false, (b), (short)0, (c), false, false)

__global__ __launch_bounds__(128)
void gemm_wmma_bf16_k(const bf16_t* __restrict__ A, int lda,
                      const bf16_t* __restrict__ B, int ldb,
                      const int* __restrict__ gather,
                      const bf16_t* __restrict__ zrow,
                      void* __restrict__ Cout, int ldc, int out_bf16,
                      int M, int N, int K,
                      const float* __restrict__ bias,
                      const float* __restrict__ bng, const float* __restrict__ bnb,
                      const float* __restrict__ bnm, const float* __restrict__ bnv,
                      int epi_bn_relu) {
  int lane = threadIdx.x & 31;
  int wave = threadIdx.x >> 5;
  int n0 = blockIdx.x * 64;
  int m0 = (blockIdx.y * 4 + wave) * 16;
  if (m0 >= M) return;
  int l15  = lane & 15;
  int hi16 = lane >> 4;

  // A fragment source row (gather<0 -> zero dummy row; loads stay unconditional)
  int arow = m0 + l15;
  const bf16_t* Ap;
  if (gather) {
    int g = gather[arow];
    Ap = (g < 0) ? zrow : A + (size_t)g * lda;
  } else {
    Ap = A + (size_t)arow * lda;
  }
  Ap += (hi16 << 4);
  const bf16_t* Bp = B + (size_t)(n0 + l15) * ldb + (hi16 << 4);
  const size_t bstr = (size_t)16 * ldb;

  v8f acc0 = {}, acc1 = {}, acc2 = {}, acc3 = {};
  for (int k = 0; k < K; k += 32) {
    v16bf a  = load_a_frag(Ap + k);
    v16bf b0 = *(const v16bf*)(Bp + k);
    v16bf b1 = *(const v16bf*)(Bp + bstr + k);
    v16bf b2 = *(const v16bf*)(Bp + 2 * bstr + k);
    v16bf b3 = *(const v16bf*)(Bp + 3 * bstr + k);
    if (k + 64 < K) {
      __builtin_prefetch(Ap + k + 64, 0, 1);
      __builtin_prefetch(Bp + k + 64, 0, 1);
      __builtin_prefetch(Bp + 3 * bstr + k + 64, 0, 1);
    }
    acc0 = WMMA_BF16(a, b0, acc0);
    acc1 = WMMA_BF16(a, b1, acc1);
    acc2 = WMMA_BF16(a, b2, acc2);
    acc3 = WMMA_BF16(a, b3, acc3);
  }

  int rbase = m0 + (hi16 << 3);
  v8f accs[4] = {acc0, acc1, acc2, acc3};
#pragma unroll
  for (int tt = 0; tt < 4; ++tt) {
    int col = n0 + tt * 16 + l15;
    if (col >= N) continue;
    float bv = bias ? bias[col] : 0.f;
    float scale = 0.f, shift = 0.f;
    if (epi_bn_relu) {
      float inv = rsqrtf(bnv[col] + 1e-5f);
      scale = bng[col] * inv;
      shift = bnb[col] - bnm[col] * scale;
    }
#pragma unroll
    for (int r = 0; r < 8; ++r) {
      float v = accs[tt][r] + bv;
      if (epi_bn_relu) v = fmaxf(fmaf(v, scale, shift), 0.f);
      size_t o = (size_t)(rbase + r) * ldc + col;
      if (out_bf16) ((bf16_t*)Cout)[o] = (bf16_t)v;
      else          ((float*)Cout)[o]  = v;
    }
  }
}

// ---------------------------------------------------------------- attention
__global__ __launch_bounds__(128)
void attention_k(const float* __restrict__ own_enc, const float* __restrict__ ctx_enc,
                 const float* __restrict__ att_w, const float* __restrict__ att_b,
                 float* __restrict__ ctx_rep) {
  int i = blockIdx.x, t = threadIdx.x;
  __shared__ float red[128];
  __shared__ float sc[K2C];
  __shared__ float s_own;
  float p = 0.f;
  for (int h = t; h < HIDN; h += 128) p += own_enc[(size_t)i * HIDN + h] * att_w[h];
  red[t] = p; __syncthreads();
  for (int s = 64; s > 0; s >>= 1) { if (t < s) red[t] += red[t + s]; __syncthreads(); }
  if (t == 0) s_own = red[0];
  __syncthreads();
  for (int j = 0; j < K2C; ++j) {
    const float* ce = ctx_enc + ((size_t)i * K2C + j) * HIDN;
    float q = 0.f;
    for (int h = t; h < HIDN; h += 128) q += ce[h] * att_w[HIDN + h];
    red[t] = q; __syncthreads();
    for (int s = 64; s > 0; s >>= 1) { if (t < s) red[t] += red[t + s]; __syncthreads(); }
    if (t == 0) sc[j] = red[0] + s_own + att_b[0];
    __syncthreads();
  }
  if (t == 0) {
    float mx = sc[0];
    for (int j = 1; j < K2C; ++j) mx = fmaxf(mx, sc[j]);
    float s = 0.f;
    for (int j = 0; j < K2C; ++j) { sc[j] = expf(sc[j] - mx); s += sc[j]; }
    for (int j = 0; j < K2C; ++j) sc[j] /= s;
  }
  __syncthreads();
  for (int h = t; h < HIDN; h += 128) {
    float a = 0.f;
    for (int j = 0; j < K2C; ++j)
      a = fmaf(sc[j], ctx_enc[((size_t)i * K2C + j) * HIDN + h], a);
    ctx_rep[(size_t)i * HIDN + h] = a;
  }
}

// ---------------------------------------------------------------- combined builder
__global__ __launch_bounds__(256)
void build_combined_k(const bf16_t* __restrict__ own_bf, const float* __restrict__ ctx_rep,
                      const float* __restrict__ bb, bf16_t* __restrict__ comb) {
  int idx = blockIdx.x * blockDim.x + threadIdx.x;
  if (idx >= NB * NTOTP) return;
  int k = idx % NTOTP, i = idx / NTOTP;
  bf16_t v;
  if (k < NVIS) v = own_bf[(size_t)i * NVIS + k];
  else if (k < NVIS + HIDN) v = (bf16_t)ctx_rep[(size_t)i * HIDN + (k - NVIS)];
  else if (k < NVIS + HIDN + 4) {
    int j = k - (NVIS + HIDN);
    float x1 = bb[i * 5 + 1], y1 = bb[i * 5 + 2];
    float x2 = bb[i * 5 + 3], y2 = bb[i * 5 + 4];
    float xv = (j == 0) ? x1 : (j == 1) ? y1 : (j == 2) ? (x2 - x1) : (y2 - y1);
    v = (bf16_t)xv;
  } else v = (bf16_t)0.f;
  comb[idx] = v;
}

// ================================================================ launch
extern "C" void kernel_launch(void* const* d_in, const int* in_sizes, int n_in,
                              void* d_out, int out_size, void* d_ws, size_t ws_size,
                              hipStream_t stream) {
  const float* images = (const float*)d_in[0];
  const float* bboxes = (const float*)d_in[1];
  const int*   ctxidx = (const int*)d_in[2];
  const float* c1w = (const float*)d_in[3];  const float* c1b = (const float*)d_in[4];
  const float* c2w = (const float*)d_in[5];  const float* c2b = (const float*)d_in[6];
  const float* c3w = (const float*)d_in[7];  const float* c3b = (const float*)d_in[8];
  const float* enc_w1 = (const float*)d_in[9];  const float* enc_b1 = (const float*)d_in[10];
  const float* enc_g  = (const float*)d_in[11]; const float* enc_be = (const float*)d_in[12];
  const float* enc_m  = (const float*)d_in[13]; const float* enc_v  = (const float*)d_in[14];
  const float* enc_w2 = (const float*)d_in[15]; const float* enc_b2 = (const float*)d_in[16];
  const float* att_w  = (const float*)d_in[17]; const float* att_b  = (const float*)d_in[18];
  const float* dec_w1 = (const float*)d_in[19]; const float* dec_b1 = (const float*)d_in[20];
  const float* dec_g  = (const float*)d_in[21]; const float* dec_be = (const float*)d_in[22];
  const float* dec_m  = (const float*)d_in[23]; const float* dec_v  = (const float*)d_in[24];
  const float* dec_w2 = (const float*)d_in[25]; const float* dec_b2 = (const float*)d_in[26];

  char* ws = (char*)d_ws;
  size_t off = 0;
  auto alloc = [&](size_t bytes) -> void* {
    void* p = ws + off;
    off = (off + bytes + 255) & ~(size_t)255;
    return p;
  };

  float*  conv1o = (float*)alloc((size_t)8 * 64 * 55 * 55 * 4);
  float*  pool1  = (float*)alloc((size_t)8 * 64 * 27 * 27 * 4);
  float*  conv2o = (float*)alloc((size_t)8 * 192 * 27 * 27 * 4);
  float*  pool2  = (float*)alloc((size_t)8 * 192 * 13 * 13 * 4);
  float*  featb  = (float*)alloc((size_t)8 * CF * 13 * 13 * 4);
  float*  own    = (float*)alloc((size_t)NB * NVIS * 4);
  bf16_t* own_bf = (bf16_t*)alloc((size_t)NB * NVIS * 2);
  bf16_t* zrow   = (bf16_t*)alloc((size_t)NVIS * 2);           // gather zero row
  bf16_t* w1T    = (bf16_t*)alloc((size_t)HIDNP64 * NVIS * 2); // 320 x 3456
  bf16_t* w2T    = (bf16_t*)alloc((size_t)HIDNP64 * HKP * 2);  // 320 x 320
  bf16_t* dw1T   = (bf16_t*)alloc((size_t)NTOTP * NTOTP * 2);  // 3776 x 3776
  bf16_t* dw2T   = (bf16_t*)alloc((size_t)64 * NTOTP * 2);     // 64 x 3776
  bf16_t* h1o    = (bf16_t*)alloc((size_t)NB * HKP * 2);
  bf16_t* h1c    = (bf16_t*)alloc((size_t)NB * K2C * HKP * 2);
  float*  oenc   = (float*)alloc((size_t)NB * HIDN * 4);
  float*  cenc   = (float*)alloc((size_t)NB * K2C * HIDN * 4);
  float*  crep   = (float*)alloc((size_t)NB * HIDN * 4);
  bf16_t* comb   = (bf16_t*)alloc((size_t)NB * NTOTP * 2);
  bf16_t* hdec   = (bf16_t*)alloc((size_t)NB * NTOTP * 2);

  auto cdiv = [](int a, int b) { return (a + b - 1) / b; };

  // ---- conv stack
  conv2d_k<<<cdiv(8 * 64 * 55 * 55, 256), 256, 0, stream>>>(
      images, c1w, c1b, conv1o, 8, 3, 224, 224, 64, 55, 55, 11, 11, 4, 2, 1);
  maxpool3s2_k<<<cdiv(8 * 64 * 27 * 27, 256), 256, 0, stream>>>(
      conv1o, pool1, 8 * 64, 55, 55, 27, 27);
  conv2d_k<<<cdiv(8 * 192 * 27 * 27, 256), 256, 0, stream>>>(
      pool1, c2w, c2b, conv2o, 8, 64, 27, 27, 192, 27, 27, 5, 5, 1, 2, 1);
  maxpool3s2_k<<<cdiv(8 * 192 * 13 * 13, 256), 256, 0, stream>>>(
      conv2o, pool2, 8 * 192, 27, 27, 13, 13);
  conv2d_k<<<cdiv(8 * CF * 13 * 13, 256), 256, 0, stream>>>(
      pool2, c3w, c3b, featb, 8, 192, 13, 13, CF, 13, 13, 3, 3, 1, 1, 0);

  // ---- ROI pool -> own (f32) -> bf16; zero gather row
  roi_pool_k<<<cdiv(NB * CF, 256), 256, 0, stream>>>(featb, bboxes, own);
  cvt_f32_bf16_k<<<cdiv(NB * NVIS, 256), 256, 0, stream>>>(own, own_bf, NB * NVIS);
  fill_bf16_zero_k<<<cdiv(NVIS, 256), 256, 0, stream>>>(zrow, NVIS);

  // ---- weight prep (transpose + bf16 + pad to ceil64 rows)
  wT_bf16_k<<<cdiv(HIDNP64 * NVIS, 256), 256, 0, stream>>>(enc_w1, w1T, NVIS, HIDN, NVIS, HIDNP64);
  wT_bf16_k<<<cdiv(HIDNP64 * HKP, 256), 256, 0, stream>>>(enc_w2, w2T, HIDN, HIDN, HKP, HIDNP64);
  wT_bf16_k<<<cdiv(NTOTP * NTOTP, 256), 256, 0, stream>>>(dec_w1, dw1T, NTOT, NTOT, NTOTP, NTOTP);
  wT_bf16_k<<<cdiv(64 * NTOTP, 256), 256, 0, stream>>>(dec_w2, dw2T, NTOT, 4, NTOTP, 64);

  // ---- zero pad regions of bf16 intermediates
  fill_bf16_zero_k<<<cdiv(NB * HKP, 256), 256, 0, stream>>>(h1o, NB * HKP);
  fill_bf16_zero_k<<<cdiv(NB * K2C * HKP, 256), 256, 0, stream>>>(h1c, NB * K2C * HKP);
  fill_bf16_zero_k<<<cdiv(NB * NTOTP, 256), 256, 0, stream>>>(hdec, NB * NTOTP);

  // ---- encoder layer 1 (BN+ReLU fused), own and gathered ctx
  {
    dim3 g(cdiv(HIDN, 64), cdiv(NB, 64));
    gemm_wmma_bf16_k<<<g, 128, 0, stream>>>(own_bf, NVIS, w1T, NVIS, nullptr, zrow,
        h1o, HKP, 1, NB, HIDN, NVIS, enc_b1, enc_g, enc_be, enc_m, enc_v, 1);
  }
  {
    dim3 g(cdiv(HIDN, 64), cdiv(NB * K2C, 64));
    gemm_wmma_bf16_k<<<g, 128, 0, stream>>>(own_bf, NVIS, w1T, NVIS, ctxidx, zrow,
        h1c, HKP, 1, NB * K2C, HIDN, NVIS, enc_b1, enc_g, enc_be, enc_m, enc_v, 1);
  }
  // ---- encoder layer 2 (bias only, f32 out)
  {
    dim3 g(cdiv(HIDN, 64), cdiv(NB, 64));
    gemm_wmma_bf16_k<<<g, 128, 0, stream>>>(h1o, HKP, w2T, HKP, nullptr, zrow,
        oenc, HIDN, 0, NB, HIDN, HKP, enc_b2, nullptr, nullptr, nullptr, nullptr, 0);
  }
  {
    dim3 g(cdiv(HIDN, 64), cdiv(NB * K2C, 64));
    gemm_wmma_bf16_k<<<g, 128, 0, stream>>>(h1c, HKP, w2T, HKP, nullptr, zrow,
        cenc, HIDN, 0, NB * K2C, HIDN, HKP, enc_b2, nullptr, nullptr, nullptr, nullptr, 0);
  }

  // ---- attention (scores + softmax + weighted sum)
  attention_k<<<NB, 128, 0, stream>>>(oenc, cenc, att_w, att_b, crep);

  // ---- combined = [own | ctx_rep | xywh | 0-pad] in bf16
  build_combined_k<<<cdiv(NB * NTOTP, 256), 256, 0, stream>>>(own_bf, crep, bboxes, comb);

  // ---- decoder layer 1 (BN+ReLU, bf16 out) and layer 2 (f32 -> d_out)
  {
    dim3 g(cdiv(NTOT, 64), cdiv(NB, 64));
    gemm_wmma_bf16_k<<<g, 128, 0, stream>>>(comb, NTOTP, dw1T, NTOTP, nullptr, zrow,
        hdec, NTOTP, 1, NB, NTOT, NTOTP, dec_b1, dec_g, dec_be, dec_m, dec_v, 1);
  }
  {
    dim3 g(1, cdiv(NB, 64));
    gemm_wmma_bf16_k<<<g, 128, 0, stream>>>(hdec, NTOTP, dw2T, NTOTP, nullptr, zrow,
        d_out, 4, 0, NB, 4, NTOTP, dec_b2, nullptr, nullptr, nullptr, nullptr, 0);
  }
}
